// MultiBoxLoss_20023137534140
// MI455X (gfx1250) — compile-verified, matrix operand order checked by reading, not run on hardware
//
#include <hip/hip_runtime.h>
#include <hip/hip_bf16.h>
#include <math.h>

// Problem constants (from the reference): B=64, A=21842, C=2, L=10
#define B_   64
#define A_   21842
#define L_   10
#define TPB  256
#define ITEMS 4
#define CHUNK (TPB * ITEMS)                         // 1024 anchors per block
#define BLOCKS_PER_ROW ((A_ + CHUNK - 1) / CHUNK)   // 22

// Pass-2 TDM staging: chunks of anchors DMA'd into LDS, double-buffered.
#define SCH    2048
#define NCHUNK ((A_ + SCH - 1) / SCH)               // 11

typedef __attribute__((ext_vector_type(8))) int v8i;
typedef __attribute__((ext_vector_type(4))) unsigned int u32x4;
typedef __attribute__((ext_vector_type(4))) int i32x4;
typedef __attribute__((ext_vector_type(8))) int i32x8;

struct Accum {
    double sl1;            // smooth-L1 numerator
    double d2;             // landmark MSE numerator
    double nll;            // selected CE numerator
    int    row_pos[B_];    // per-row positive counts
    unsigned pivot[B_];    // k-th largest mining key (sortable-uint)
    unsigned istar[B_];    // index cutoff among key==pivot ties
    int    num_pos;        // total positives
    int    n_sel;          // total selected anchors
};

// float -> order-preserving uint (larger float => larger uint)
__device__ inline unsigned sortable(float f) {
    unsigned u = __float_as_uint(f);
    return (u & 0x80000000u) ? ~u : (u | 0x80000000u);
}

// Mining key from the two logits + label.  noinline => bit-identical result
// whether the operands came from LDS (pass2) or global (pass3).
__device__ __attribute__((noinline))
unsigned key_from(float x0, float x1, int y) {
    float mx = fmaxf(x0, x1), mn = fminf(x0, x1);
    float lse = mx + log1pf(expf(mn - mx));          // stable 2-class LSE
    float ce  = lse - (y > 0 ? x1 : x0);
    float mining = (y > 0) ? 0.0f : ce;
    return sortable(mining);
}

// Exact wave-wide popcount of 4 flags/lane via V_WMMA_I32_16X16X64_IU8.
// B == all-ones bytes, so sum_m D[m][n] over one column == total flag count
// regardless of the A byte mapping.  Column 0 lives in lane 0 (M=0..7) and
// lane 16 (M=8..15) across the 8 D VGPRs.
__device__ inline int wave_count_flags(int f0, int f1, int f2, int f3) {
    v8i a = {0, 0, 0, 0, 0, 0, 0, 0};
    a[0] = (f0 & 1) | ((f1 & 1) << 8) | ((f2 & 1) << 16) | ((f3 & 1) << 24);
    v8i ones = {0x01010101, 0x01010101, 0x01010101, 0x01010101,
                0x01010101, 0x01010101, 0x01010101, 0x01010101};
    v8i c = {0, 0, 0, 0, 0, 0, 0, 0};
    v8i d = __builtin_amdgcn_wmma_i32_16x16x64_iu8(false, a, false, ones, c,
                                                   false, false);
    int s = d[0] + d[1] + d[2] + d[3] + d[4] + d[5] + d[6] + d[7];
    return __shfl(s, 0, 32) + __shfl(s, 16, 32);
}

__device__ inline float wave_sum(float v) {
    for (int o = 16; o > 0; o >>= 1) v += __shfl_down(v, o, 32);
    return v;   // valid in lane 0
}

// Flat address of a __shared__ object -> 32-bit LDS byte address (LDS
// aperture maps the low 32 bits to the LDS offset).
__device__ inline unsigned lds_addr_of(const void* p) {
    return (unsigned)(unsigned long long)p;
}

// Issue one TDM 1-D tile load: nelem 4-byte elements from global -> LDS.
// D# built per CDNA5 ISA ch.8: group0 {count=1, lds_addr, global_addr, type=2},
// group1 {data_size=4B, tensor_dim0=tile_dim0=nelem, tensor_dim1=1,
// tensor_dim0_stride=nelem, workgroup_mask=0}, groups 2/3 zero (<=2D tensor).
// This toolchain's builtin is the 6-arg form:
//   (u32x4 g0, i32x8 g1, i32x4, i32x4, i32x8, i32 cpol)
__device__ inline void tdm_load_1d(const void* g, unsigned lds, int nelem) {
    unsigned long long ga = (unsigned long long)g;
    u32x4 g0 = {0u, 0u, 0u, 0u};
    g0[0] = 1u;                                   // count=1 (valid user D#)
    g0[1] = lds;                                  // lds_addr
    g0[2] = (unsigned)ga;                         // global_addr[31:0]
    g0[3] = (unsigned)((ga >> 32) & 0x1FFFFFFull) // global_addr[56:32]
          | 0x80000000u;                          // type=2 ("image")
    i32x8 g1 = {0, 0, 0, 0, 0, 0, 0, 0};
    g1[0] = (2 << 16);                            // data_size=4B; wg_mask=0
    g1[1] = (nelem & 0xFFFF) << 16;               // tensor_dim0[15:0]
    g1[2] = ((nelem >> 16) & 0xFFFF)              // tensor_dim0[31:16]
          | (1 << 16);                            // tensor_dim1 = 1
    g1[3] = (nelem & 0xFFFF) << 16;               // tile_dim0 (<=65535)
    g1[5] = nelem;                                // tensor_dim0_stride[31:0]
    i32x4 z4 = {0, 0, 0, 0};
    i32x8 z8 = {0, 0, 0, 0, 0, 0, 0, 0};
    __builtin_amdgcn_tensor_load_to_lds(g0, g1, z4, z4, z8, 0);
}

__global__ void init_ws(Accum* acc) {
    int t = threadIdx.x;
    if (t == 0) {
        acc->sl1 = 0.0; acc->d2 = 0.0; acc->nll = 0.0;
        acc->num_pos = 0; acc->n_sel = 0;
    }
    if (t < B_) acc->row_pos[t] = 0;
}

// Pass 1: stream loc/landmark tensors once (HBM-bound, ~145 MB); smooth-L1 +
// MSE partials and positive counts (counts via integer WMMA popcount).
__global__ void pass1(const float* __restrict__ lp, const float* __restrict__ lt,
                      const float* __restrict__ mp, const float* __restrict__ mt,
                      const int* __restrict__ tg, Accum* acc) {
    int row = blockIdx.x / BLOCKS_PER_ROW;
    int c0  = (blockIdx.x % BLOCKS_PER_ROW) * CHUNK;
    int t   = threadIdx.x;
    float sl1 = 0.f, d2 = 0.f;
    int fl[ITEMS];
    #pragma unroll
    for (int i = 0; i < ITEMS; i++) {
        int col = c0 + i * TPB + t;
        int pos = 0;
        if (col < A_) {
            long a = (long)row * A_ + col;
            pos = tg[a] > 0;
            if (pos) {
                long a4 = a * 4, aL = a * L_;
                #pragma unroll
                for (int j = 0; j < 4; j++) {
                    float d  = lp[a4 + j] - lt[a4 + j];
                    float ad = fabsf(d);
                    sl1 += (ad < 1.f) ? 0.5f * d * d : ad - 0.5f;
                }
                #pragma unroll
                for (int j = 0; j < L_; j++) {
                    float d = mp[aL + j] - mt[aL + j];
                    d2 += d * d;
                }
            }
        }
        fl[i] = pos;
    }
    int cnt  = wave_count_flags(fl[0], fl[1], fl[2], fl[3]);   // EXEC all-1s
    float s1 = wave_sum(sl1);
    float s2 = wave_sum(d2);
    if ((t & 31) == 0) {
        atomicAdd(&acc->sl1, (double)s1);
        atomicAdd(&acc->d2,  (double)s2);
        if (cnt) {
            atomicAdd(&acc->row_pos[row], cnt);
            atomicAdd(&acc->num_pos, cnt);
        }
    }
}

// Pass 2: per-row radix select of the k-th largest mining key + tie cutoff.
// All element data is staged into LDS by the Tensor Data Mover with double
// buffering: wave 0 issues TENSOR_LOAD_TO_LDS for chunk c+1, waits
// TENSORcnt<=2 (chunk c complete), barrier, everyone processes chunk c from
// LDS while the TDM fills the other buffer.
__global__ void pass2(const float* __restrict__ cp, const int* __restrict__ tg,
                      Accum* acc) {
    int row = blockIdx.x;
    int t   = threadIdx.x;
    __shared__ float     s_conf[2][SCH * 2];   // 32 KB
    __shared__ int       s_tg[2][SCH];         // 16 KB
    __shared__ int       hist[256];
    __shared__ unsigned  sh_prefix;
    __shared__ int       sh_t;
    __shared__ unsigned  sh_mask[TPB / 32];
    __shared__ int       sh_done;
    __shared__ unsigned  sh_istar;

    int np = acc->row_pos[row];
    int k  = 3 * np; if (k > A_ - 1) k = A_ - 1;
    if (k <= 0) {
        if (t == 0) { acc->pivot[row] = 0xFFFFFFFFu; acc->istar[row] = 0u; }
        return;
    }
    if (t == 0) { sh_prefix = 0u; sh_t = k; }

    const float* rowc = cp + (long)row * A_ * 2;
    const int*   rowt = tg + (long)row * A_;

    unsigned maskhi = 0u;
    for (int p = 0; p < 4; p++) {            // MSB-first 8-bit digits
        int shift = 24 - 8 * p;
        hist[t] = 0;
        if (t < 32) {                         // wave 0: stage chunk 0
            int n0 = (A_ < SCH) ? A_ : SCH;
            tdm_load_1d(rowc, lds_addr_of(&s_conf[0][0]), 2 * n0);
            tdm_load_1d(rowt, lds_addr_of(&s_tg[0][0]),   n0);
        }
        __syncthreads();
        unsigned prefix = sh_prefix;
        for (int c = 0; c < NCHUNK; c++) {
            int base = c * SCH;
            int n_c  = A_ - base; if (n_c > SCH) n_c = SCH;
            if (t < 32) {
                if (c + 1 < NCHUNK) {         // prefetch next chunk via TDM
                    int nb = base + SCH;
                    int nn = A_ - nb; if (nn > SCH) nn = SCH;
                    int b  = (c + 1) & 1;
                    tdm_load_1d(rowc + 2L * nb, lds_addr_of(&s_conf[b][0]), 2 * nn);
                    tdm_load_1d(rowt + nb,      lds_addr_of(&s_tg[b][0]),   nn);
                    __builtin_amdgcn_s_wait_tensorcnt(2);  // chunk c landed
                } else {
                    __builtin_amdgcn_s_wait_tensorcnt(0);
                }
            }
            __syncthreads();
            int b = c & 1;
            for (int i = t; i < n_c; i += TPB) {
                unsigned key = key_from(s_conf[b][2 * i], s_conf[b][2 * i + 1],
                                        s_tg[b][i]);
                if ((key & maskhi) == prefix)
                    atomicAdd(&hist[(key >> shift) & 0xFFu], 1);
            }
            __syncthreads();                  // buffer b reusable afterwards
        }
        if (t == 0) {
            int rem = sh_t, run = 0, digit = 0;
            for (int i = 255; i >= 0; i--) {          // descending scan
                if (run + hist[i] >= rem) { digit = i; rem -= run; break; }
                run += hist[i];
            }
            sh_prefix = prefix | ((unsigned)digit << shift);
            sh_t = rem;                                // rank within pivot bin
        }
        __syncthreads();
        maskhi |= 0xFFu << shift;
    }

    unsigned pivot = sh_prefix;
    int T = sh_t;                 // #(key==pivot) elements to take (>=1)
    if (t == 0) { sh_done = 0; sh_istar = 0; }
    int lane = t & 31, wid = t >> 5;
    int running = 0;              // meaningful in thread 0 only

    if (t < 32) {                 // stage chunk 0 for the ordered tie scan
        int n0 = (A_ < SCH) ? A_ : SCH;
        tdm_load_1d(rowc, lds_addr_of(&s_conf[0][0]), 2 * n0);
        tdm_load_1d(rowt, lds_addr_of(&s_tg[0][0]),   n0);
    }
    __syncthreads();
    for (int c = 0; c < NCHUNK; c++) {
        int base = c * SCH;
        int n_c  = A_ - base; if (n_c > SCH) n_c = SCH;
        if (t < 32) {
            if (c + 1 < NCHUNK) {
                int nb = base + SCH;
                int nn = A_ - nb; if (nn > SCH) nn = SCH;
                int b  = (c + 1) & 1;
                tdm_load_1d(rowc + 2L * nb, lds_addr_of(&s_conf[b][0]), 2 * nn);
                tdm_load_1d(rowt + nb,      lds_addr_of(&s_tg[b][0]),   nn);
                __builtin_amdgcn_s_wait_tensorcnt(2);
            } else {
                __builtin_amdgcn_s_wait_tensorcnt(0);
            }
        }
        __syncthreads();
        int b = c & 1;
        for (int s2 = 0; s2 < SCH; s2 += TPB) {   // uniform trip count
            int i = s2 + t;
            int eq = 0;
            if (i < n_c)
                eq = (key_from(s_conf[b][2 * i], s_conf[b][2 * i + 1],
                               s_tg[b][i]) == pivot);
            unsigned long long bal = __ballot(eq);
            if (lane == 0) sh_mask[wid] = (unsigned)bal;
            __syncthreads();
            if (t == 0 && !sh_done) {
                int cc = 0;
                for (int w = 0; w < TPB / 32; w++) cc += __popc(sh_mask[w]);
                if (running + cc >= T) {
                    int need = T - running, istar = 0;
                    for (int w = 0; w < TPB / 32 && need > 0; w++) {
                        unsigned m = sh_mask[w];
                        while (m && need > 0) {
                            int bit = __ffs(m) - 1;
                            m &= m - 1;
                            if (--need == 0) istar = base + s2 + w * 32 + bit + 1;
                        }
                    }
                    sh_istar = (unsigned)istar;
                    sh_done = 1;
                }
                running += cc;
            }
            __syncthreads();
        }
        if (sh_done) break;       // uniform exit
    }
    if (t < 32) __builtin_amdgcn_s_wait_tensorcnt(0);   // drain pending TDM
    __syncthreads();
    if (t == 0) { acc->pivot[row] = pivot; acc->istar[row] = sh_istar; }
}

// Pass 3: selected-CE sum + selected count (count via WMMA popcount).
__global__ void pass3(const float* __restrict__ cp, const int* __restrict__ tg,
                      Accum* acc) {
    int row = blockIdx.x / BLOCKS_PER_ROW;
    int c0  = (blockIdx.x % BLOCKS_PER_ROW) * CHUNK;
    int t   = threadIdx.x;
    unsigned pivot = acc->pivot[row];
    unsigned istar = acc->istar[row];
    float nll = 0.f;
    int fl[ITEMS];
    #pragma unroll
    for (int i = 0; i < ITEMS; i++) {
        int col = c0 + i * TPB + t;
        int sel = 0;
        if (col < A_) {
            long a = (long)row * A_ + col;
            int   y  = tg[a];
            float x0 = cp[2 * a], x1 = cp[2 * a + 1];
            float mx = fmaxf(x0, x1), mn = fminf(x0, x1);
            float ce = mx + log1pf(expf(mn - mx)) - (y > 0 ? x1 : x0);
            unsigned key = key_from(x0, x1, y);
            sel = (y > 0) || (key > pivot) ||
                  (key == pivot && (unsigned)col < istar);
            if (sel) nll += ce;
        }
        fl[i] = sel;
    }
    int cnt = wave_count_flags(fl[0], fl[1], fl[2], fl[3]);   // EXEC all-1s
    float s = wave_sum(nll);
    if ((t & 31) == 0) {
        atomicAdd(&acc->nll, (double)s);
        if (cnt) atomicAdd(&acc->n_sel, cnt);
    }
}

__global__ void finalize_k(const Accum* acc, float* out) {
    if (threadIdx.x == 0 && blockIdx.x == 0) {
        double np = (double)acc->num_pos;
        out[0] = (float)(acc->sl1 / (np * 4.0));
        out[1] = (float)(acc->d2  / (np * 10.0));
        out[2] = (float)(acc->nll / (double)acc->n_sel);
    }
}

extern "C" void kernel_launch(void* const* d_in, const int* in_sizes, int n_in,
                              void* d_out, int out_size, void* d_ws, size_t ws_size,
                              hipStream_t stream) {
    (void)in_sizes; (void)n_in; (void)out_size; (void)ws_size;
    const float* lp = (const float*)d_in[0];   // loc_preds            [B,A,4]
    const float* lt = (const float*)d_in[1];   // loc_targets          [B,A,4]
    const float* mp = (const float*)d_in[2];   // landmarks_preds      [B,A,10]
    const float* mt = (const float*)d_in[3];   // landmarks_targets    [B,A,10]
    const float* cp = (const float*)d_in[4];   // conf_preds           [B,A,2]
    const int*   tg = (const int*)d_in[5];     // conf_targets         [B,A]
    float* out = (float*)d_out;                // (loc, landmarks, conf) losses
    Accum* acc = (Accum*)d_ws;                 // < 1 KB scratch

    init_ws<<<1, 64, 0, stream>>>(acc);
    dim3 grid(B_ * BLOCKS_PER_ROW);            // 1408 blocks
    pass1<<<grid, TPB, 0, stream>>>(lp, lt, mp, mt, tg, acc);
    pass2<<<B_, TPB, 0, stream>>>(cp, tg, acc);
    pass3<<<grid, TPB, 0, stream>>>(cp, tg, acc);
    finalize_k<<<1, 1, 0, stream>>>(acc, out);
}
